// FluxonRouter_15444702396965
// MI455X (gfx1250) — matrix operand order repaired and checked by reading, not compiled, not run on hardware
//
#include <hip/hip_runtime.h>

typedef __attribute__((ext_vector_type(16))) _Float16 v16h;
typedef __attribute__((ext_vector_type(8)))  float    v8f;

#define FLX_B   131072
#define FLX_IN  1024
#define FLX_S   256
#define FLX_NF  64
#define FLX_TAU 1.9985f

// ---------------------------------------------------------------------------
// Kernel 0a: Kmat[f][s] = sum_j A[f][j] * Wk[s][j]          (64 x 256)
// ---------------------------------------------------------------------------
__global__ __launch_bounds__(256) void flx_kmat(const float* __restrict__ A,
                                                const float* __restrict__ Wk,
                                                float* __restrict__ Kmat) {
  const int f = blockIdx.x;     // 0..63
  const int s = threadIdx.x;    // 0..255
  const float* ar = A  + f * FLX_S;
  const float* wr = Wk + s * FLX_S;
  float acc = 0.f;
  for (int j = 0; j < FLX_S; ++j) acc = fmaf(ar[j], wr[j], acc);
  Kmat[f * FLX_S + s] = acc;
}

// ---------------------------------------------------------------------------
// Kernel 0b: M[d][f] = (1/TAU) * sum_s Wq[s][d] * Kmat[f][s]   (1024 x 64)
// stored as f16 pre-swizzled into WMMA B-matrix (32x16, 16-bit) lane layout:
//   flat = ((kstep*4 + ntile)*32 + lane)*16 + half
//   lane = ((K>>3)&1)*16 + (f&15),  half = (K&7) | (((K&31)>>4)<<3),  ntile = f>>4
// ---------------------------------------------------------------------------
__global__ __launch_bounds__(256) void flx_mpack(const float* __restrict__ Wq,
                                                 const float* __restrict__ Kmat,
                                                 _Float16* __restrict__ Bpk) {
  const int gid = blockIdx.x * 256 + threadIdx.x;  // 0..65535
  const int d = gid >> 6;     // 0..1023  (K dimension)
  const int f = gid & 63;     // 0..63    (N dimension)
  const float* kr = Kmat + f * FLX_S;
  float acc = 0.f;
  for (int s = 0; s < FLX_S; ++s) acc = fmaf(Wq[s * FLX_IN + d], kr[s], acc);
  acc *= (1.0f / FLX_TAU);
  const int ks   = d >> 5;
  const int kk   = d & 31;
  const int lane = (((kk >> 3) & 1) << 4) | (f & 15);
  const int half = (kk & 7) | ((kk >> 4) << 3);
  const int t    = f >> 4;
  Bpk[((((ks * 4 + t) * 32) + lane) << 4) + half] = (_Float16)acc;
}

// XOR-swizzled LDS index (16B-chunk granularity): conflict-free row reads,
// keeps float4 alignment, total exactly 64 KB for 256 rows x 64 cols.
__device__ __forceinline__ int scIdx(int row, int col) {
  return row * 64 + ((((col >> 2) ^ (row & 15)) << 2) | (col & 3));
}

// ---------------------------------------------------------------------------
// Kernel 1: scores = h @ M via WMMA f16, fused entmax1.5 + top-3 epilogue.
// 256 threads = 8 waves; wave w handles 32 rows (two 16-row A tiles sharing
// each B fragment) -> 256 rows per block, one row per thread in the epilogue.
// ---------------------------------------------------------------------------
__global__ __launch_bounds__(256) void fluxon_main(const float* __restrict__ h,
                                                   const _Float16* __restrict__ Bpk,
                                                   float* __restrict__ outp) {
  __shared__ float sc[256 * 64];   // 65536 B, XOR-swizzled

  const int wave = threadIdx.x >> 5;   // 0..7
  const int lane = threadIdx.x & 31;
  const int lhi  = lane >> 4;          // 0/1
  const int llo  = lane & 15;

  const long rowbase = (long)blockIdx.x * 256 + wave * 32;
  const float* hrow0 = h + (rowbase + llo) * (long)FLX_IN;
  const float* hrow1 = hrow0 + 16 * (long)FLX_IN;

  v8f acc0[4] = {v8f{}, v8f{}, v8f{}, v8f{}};
  v8f acc1[4] = {v8f{}, v8f{}, v8f{}, v8f{}};

  for (int ks = 0; ks < 32; ++ks) {
    const int k0 = ks * 32;
    // A fragments (16-bit A 16x32 layout): lane<16: halves 0..7 -> K 0..7,
    // halves 8..15 -> K 16..23; lanes>=16: +8 on both segments.
    const float4 a0 = *(const float4*)(hrow0 + k0 + lhi * 8);
    const float4 a1 = *(const float4*)(hrow0 + k0 + lhi * 8 + 4);
    const float4 a2 = *(const float4*)(hrow0 + k0 + 16 + lhi * 8);
    const float4 a3 = *(const float4*)(hrow0 + k0 + 16 + lhi * 8 + 4);
    const float4 b0 = *(const float4*)(hrow1 + k0 + lhi * 8);
    const float4 b1 = *(const float4*)(hrow1 + k0 + lhi * 8 + 4);
    const float4 b2 = *(const float4*)(hrow1 + k0 + 16 + lhi * 8);
    const float4 b3 = *(const float4*)(hrow1 + k0 + 16 + lhi * 8 + 4);

    union { v16h v; _Float16 e[16]; } A0, A1;
    A0.e[0]  = (_Float16)a0.x;  A0.e[1]  = (_Float16)a0.y;
    A0.e[2]  = (_Float16)a0.z;  A0.e[3]  = (_Float16)a0.w;
    A0.e[4]  = (_Float16)a1.x;  A0.e[5]  = (_Float16)a1.y;
    A0.e[6]  = (_Float16)a1.z;  A0.e[7]  = (_Float16)a1.w;
    A0.e[8]  = (_Float16)a2.x;  A0.e[9]  = (_Float16)a2.y;
    A0.e[10] = (_Float16)a2.z;  A0.e[11] = (_Float16)a2.w;
    A0.e[12] = (_Float16)a3.x;  A0.e[13] = (_Float16)a3.y;
    A0.e[14] = (_Float16)a3.z;  A0.e[15] = (_Float16)a3.w;
    A1.e[0]  = (_Float16)b0.x;  A1.e[1]  = (_Float16)b0.y;
    A1.e[2]  = (_Float16)b0.z;  A1.e[3]  = (_Float16)b0.w;
    A1.e[4]  = (_Float16)b1.x;  A1.e[5]  = (_Float16)b1.y;
    A1.e[6]  = (_Float16)b1.z;  A1.e[7]  = (_Float16)b1.w;
    A1.e[8]  = (_Float16)b2.x;  A1.e[9]  = (_Float16)b2.y;
    A1.e[10] = (_Float16)b2.z;  A1.e[11] = (_Float16)b2.w;
    A1.e[12] = (_Float16)b3.x;  A1.e[13] = (_Float16)b3.y;
    A1.e[14] = (_Float16)b3.z;  A1.e[15] = (_Float16)b3.w;

#pragma unroll
    for (int t = 0; t < 4; ++t) {
      const v16h Bv = *(const v16h*)(Bpk + ((((ks * 4 + t) * 32) + lane) << 4));
      acc0[t] = __builtin_amdgcn_wmma_f32_16x16x32_f16(
          false, A0.v, false, Bv, (short)0, acc0[t], false, false);
      acc1[t] = __builtin_amdgcn_wmma_f32_16x16x32_f16(
          false, A1.v, false, Bv, (short)0, acc1[t], false, false);
    }
  }

  // Scatter C (16x16 f32 layout: VGPR v -> row v (lanes<16) / v+8 (lanes>=16),
  // col = lane&15) into swizzled LDS.
#pragma unroll
  for (int t = 0; t < 4; ++t) {
#pragma unroll
    for (int v = 0; v < 8; ++v) {
      const int r0 = (wave << 5) + (lhi ? v + 8 : v);
      const int c  = (t << 4) + llo;
      sc[scIdx(r0, c)]      = acc0[t][v];
      sc[scIdx(r0 + 16, c)] = acc1[t][v];
    }
  }
  __syncthreads();

  // ------------------- entmax-1.5 + stable top-3 per row -------------------
  {
    const int rl = threadIdx.x;                 // one row per thread
    const long grow = (long)blockIdx.x * 256 + rl;

    float x[64];
#pragma unroll
    for (int i = 0; i < 64; ++i) x[i] = sc[scIdx(rl, i)];

    float m = x[0];
#pragma unroll
    for (int i = 1; i < 64; ++i) m = fmaxf(m, x[i]);
#pragma unroll
    for (int i = 0; i < 64; ++i) x[i] -= m;

    // Expansion: f(-1) >= 0 always (max term contributes 1). If f(-1) <= 0,
    // one doubling to -2 suffices since f(-2) >= 3 > 0 (max term gives 4).
    float fl = -1.0f;
#pragma unroll
    for (int i = 0; i < 64; ++i) {
      float d = fmaxf(x[i] + 1.0f, 0.f);
      fl = fmaf(d, d, fl);
    }
    float l = (fl <= 0.f) ? -2.0f : -1.0f;
    float r = 0.f;

    for (int it = 0; it < 50; ++it) {
      const float mid = 0.5f * (l + r);
      float fm = -1.0f;
#pragma unroll
      for (int i = 0; i < 64; ++i) {
        float d = fmaxf(x[i] - mid, 0.f);
        fm = fmaf(d, d, fm);
      }
      if (fm > 0.f) l = mid; else r = mid;
    }
    const float tau = 0.5f * (l + r);

    // Stable top-3 on p = relu(x - tau)^2 (strict >, ascending index ==
    // lax.top_k tie-breaking). Global prob normalizer cancels in weights.
    float v1 = -1.f, v2 = -1.f, v3 = -1.f;
    int   i1 = 0,    i2 = 0,    i3 = 0;
#pragma unroll
    for (int i = 0; i < 64; ++i) {
      float d = fmaxf(x[i] - tau, 0.f);
      float p = d * d;
      if (p > v1)      { v3 = v2; i3 = i2; v2 = v1; i2 = i1; v1 = p; i1 = i; }
      else if (p > v2) { v3 = v2; i3 = i2; v2 = p;  i2 = i; }
      else if (p > v3) { v3 = p;  i3 = i; }
    }
    const float inv = 1.0f / (v1 + v2 + v3 + 1e-12f);

    int* oi = (int*)outp;
    oi[grow * 3 + 0] = i1;
    oi[grow * 3 + 1] = i2;
    oi[grow * 3 + 2] = i3;
    float* ow = outp + (long)FLX_B * 3;
    ow[grow * 3 + 0] = v1 * inv;
    ow[grow * 3 + 1] = v2 * inv;
    ow[grow * 3 + 2] = v3 * inv;
    if (grow == 0) outp[(long)FLX_B * 6] = FLX_TAU;
  }
}

// ---------------------------------------------------------------------------
extern "C" void kernel_launch(void* const* d_in, const int* in_sizes, int n_in,
                              void* d_out, int out_size, void* d_ws, size_t ws_size,
                              hipStream_t stream) {
  const float* h_concat = (const float*)d_in[0];   // (131072, 1024)
  const float* A_states = (const float*)d_in[1];   // (64, 256)
  const float* Wq       = (const float*)d_in[2];   // (256, 1024)
  const float* Wk       = (const float*)d_in[3];   // (256, 256)
  float* outp = (float*)d_out;

  float*     Kmat = (float*)d_ws;                        // 64 KB
  _Float16*  Bpk  = (_Float16*)((char*)d_ws + 65536);    // 128 KB

  flx_kmat <<<FLX_NF, 256, 0, stream>>>(A_states, Wk, Kmat);
  flx_mpack<<<256,    256, 0, stream>>>(Wq, Kmat, Bpk);
  fluxon_main<<<FLX_B / 256, 256, 0, stream>>>(h_concat, Bpk, outp);
}
